// Model_1279900254943
// MI455X (gfx1250) — matrix operand order repaired and checked by reading, not compiled, not run on hardware
//
#include <hip/hip_runtime.h>
#include <hip/hip_bf16.h>
#include <math.h>

typedef __attribute__((ext_vector_type(16))) _Float16 v16h;
typedef __attribute__((ext_vector_type(8)))  float    v8f;
typedef __attribute__((ext_vector_type(2)))  _Float16 h2;

#define B_  4
#define N_  512
#define D_  64
#define K_  256   // pooled nodes (N_ * 0.5)

// CDNA5 hardware tanh (V_TANH_F32, transcendental unit) — branch-free,
// replaces libocml's branchy ~30-op expansion.
__device__ __forceinline__ float fast_tanh(float x) {
#if __has_builtin(__builtin_amdgcn_tanhf)
    return __builtin_amdgcn_tanhf(x);
#else
    float y;
    asm volatile("v_tanh_f32 %0, %1\n\tv_nop" : "=v"(y) : "v"(x));
    return y;
#endif
}

__device__ __forceinline__ float selu_f(float x) {
    const float scale = 1.0507009873554805f;
    const float alpha = 1.6732632423543772f;
    return scale * (x > 0.f ? x : alpha * (__expf(x) - 1.f));
}

// ---------------------------------------------------------------------------
// K1: fused pairwise score computation (the 8.6 GFLOP GEMM) using WMMA f16.
// scores_t[b][j][i] = sum_n attw[n] * tanh( sum_d x[b,i,d]*x[b,j,d]*W[d,n] + ba[n] )
// One wave per 16i x 16j tile. Block = 4 waves = 4 consecutive j-tiles.
// ---------------------------------------------------------------------------
__global__ __launch_bounds__(128)
void k1_scores(const float* __restrict__ x, const float* __restrict__ Wa,
               const float* __restrict__ ba, const float* __restrict__ attw,
               float* __restrict__ scores_t)
{
    __shared__ __align__(32) _Float16 ldsB[8][32][16]; // [n_tile*2+kk][lane][half], B-matrix layout
    __shared__ __align__(4)  _Float16 xi[16][64];      // f16 x rows for this i-tile
    __shared__ __align__(4)  _Float16 xj[64][64];      // f16 x rows for 4 j-tiles

    const int tid = threadIdx.x;
    const int bid = blockIdx.x;          // B_ * 32 * 8 = 1024 blocks
    const int b       = bid >> 8;
    const int rem     = bid & 255;
    const int i_tile  = rem >> 3;
    const int j_group = rem & 7;
    const int i0  = i_tile * 16;
    const int jg0 = j_group * 64;

    // Pack att_proj_w (row-major [d][n]) into WMMA B-operand layout:
    // lane L: N = nt*16 + (L&15); half h -> K = kk*32 + (h<8 ? h : h+8) + ((L>=16)?8:0)
    for (int e = tid; e < 8 * 32 * 16; e += 128) {
        int frag = e >> 9, lane = (e >> 4) & 31, hh = e & 15;
        int nt = frag >> 1, kk = frag & 1;
        int Nn = nt * 16 + (lane & 15);
        int Kk = kk * 32 + ((hh < 8) ? hh : (hh + 8)) + ((lane >> 4) << 3);
        ldsB[frag][lane][hh] = (_Float16)Wa[Kk * 64 + Nn];
    }
    for (int e = tid; e < 16 * 64; e += 128) {
        int r = e >> 6, d = e & 63;
        xi[r][d] = (_Float16)x[((size_t)(b * N_ + i0 + r)) * D_ + d];
    }
    for (int e = tid; e < 64 * 64; e += 128) {
        int r = e >> 6, d = e & 63;
        xj[r][d] = (_Float16)x[((size_t)(b * N_ + jg0 + r)) * D_ + d];
    }
    __syncthreads();

    const int wave = tid >> 5;
    const int lane = tid & 31;
    const int nlo  = lane & 15;            // N index within tile (C layout), also j-row selector
    const int hiA  = (lane >> 4) << 3;     // +8 K-offset for upper half-wave (A layout)
    const int j0   = jg0 + wave * 16;

    float bias_n[4], aw_n[4];
#pragma unroll
    for (int nt = 0; nt < 4; ++nt) {
        bias_n[nt] = ba[nt * 16 + nlo];
        aw_n[nt]   = attw[nt * 16 + nlo];
    }

    // Preload this lane's x_j packed pairs (A layout K pairs), reused for all 16 i.
    // pair v: K0 = kk*32 + (v<4 ? 2v : 2v+8) + hiA
    h2 xjp[2][8];
    const _Float16* xjr = &xj[wave * 16 + nlo][0];
#pragma unroll
    for (int kk = 0; kk < 2; ++kk)
#pragma unroll
        for (int v = 0; v < 8; ++v) {
            int K0 = kk * 32 + ((v < 4) ? (2 * v) : (2 * v + 8)) + hiA;
            xjp[kk][v] = *(const h2*)(xjr + K0);
        }

    for (int il = 0; il < 16; ++il) {
        // Build A fragments: A[M=nlo][K] = x_i[K] * x_j[K]  (packed f16 mul)
        union { v16h v; h2 p[8]; } A[2];
        const _Float16* xir = &xi[il][0];
#pragma unroll
        for (int kk = 0; kk < 2; ++kk)
#pragma unroll
            for (int v = 0; v < 8; ++v) {
                int K0 = kk * 32 + ((v < 4) ? (2 * v) : (2 * v + 8)) + hiA;
                h2 a = *(const h2*)(xir + K0);
                A[kk].p[v] = a * xjp[kk][v];
            }

        float sc[8];
#pragma unroll
        for (int r = 0; r < 8; ++r) sc[r] = 0.f;

#pragma unroll
        for (int nt = 0; nt < 4; ++nt) {
            v8f acc;
#pragma unroll
            for (int r = 0; r < 8; ++r) acc[r] = bias_n[nt];  // bias folded into accumulator
            v16h b0 = *(const v16h*)&ldsB[nt * 2 + 0][lane][0];
            v16h b1 = *(const v16h*)&ldsB[nt * 2 + 1][lane][0];
            acc = __builtin_amdgcn_wmma_f32_16x16x32_f16(false, A[0].v, false, b0,
                                                         (short)0, acc, false, false);
            acc = __builtin_amdgcn_wmma_f32_16x16x32_f16(false, A[1].v, false, b1,
                                                         (short)0, acc, false, false);
#pragma unroll
            for (int r = 0; r < 8; ++r)
                sc[r] += aw_n[nt] * fast_tanh(acc[r]);
        }

        // Reduce over N (16 lanes per half-wave). Half-waves hold disjoint j rows.
#pragma unroll
        for (int m = 8; m >= 1; m >>= 1)
#pragma unroll
            for (int r = 0; r < 8; ++r)
                sc[r] += __shfl_xor(sc[r], m, 32);

        if ((lane & 15) == 0) {
            const int jb = j0 + hiA;          // lane 0 -> j_local 0..7, lane 16 -> 8..15
            const int i  = i0 + il;
#pragma unroll
            for (int r = 0; r < 8; ++r)
                scores_t[((size_t)(b * N_) + jb + r) * N_ + i] = sc[r];
        }
    }
}

// ---------------------------------------------------------------------------
// K2: softmax over i (contiguous rows of scores_t), in place.
// ---------------------------------------------------------------------------
__global__ __launch_bounds__(128)
void k2_softmax(float* __restrict__ s)
{
    __shared__ float red[128];
    const int row = blockIdx.x;              // b*N_ + j
    float* p = s + (size_t)row * N_;
    const int t = threadIdx.x;

    float v[4];
    float mx = -3.4e38f;
#pragma unroll
    for (int q = 0; q < 4; ++q) { v[q] = p[t + 128 * q]; mx = fmaxf(mx, v[q]); }
    red[t] = mx; __syncthreads();
    for (int o = 64; o >= 1; o >>= 1) {
        if (t < o) red[t] = fmaxf(red[t], red[t + o]);
        __syncthreads();
    }
    mx = red[0]; __syncthreads();

    float sm = 0.f;
#pragma unroll
    for (int q = 0; q < 4; ++q) { v[q] = __expf(v[q] - mx); sm += v[q]; }
    red[t] = sm; __syncthreads();
    for (int o = 64; o >= 1; o >>= 1) {
        if (t < o) red[t] += red[t + o];
        __syncthreads();
    }
    const float inv = 1.f / red[0];
#pragma unroll
    for (int q = 0; q < 4; ++q) p[t + 128 * q] = v[q] * inv;
}

// ---------------------------------------------------------------------------
// K3: agg[b,i,:] = sum_j att_t[b,j,i] * x[b,j,:] ; h = agg@W1 + b1 + x@W2 + b2
// Block = 256 threads = 4 rows (i), 64 channels each.
// ---------------------------------------------------------------------------
__global__ __launch_bounds__(256)
void k3_agg_h(const float* __restrict__ x, const float* __restrict__ att_t,
              const float* __restrict__ W1, const float* __restrict__ b1,
              const float* __restrict__ W2, const float* __restrict__ b2,
              float* __restrict__ h)
{
    __shared__ float agg[4][64];
    __shared__ float xr[4][64];
    const int t = threadIdx.x;
    const int c = t & 63, g = t >> 6;
    const int gid = blockIdx.x;              // B_ * 128 = 512 blocks
    const int b = gid >> 7;
    const int i = ((gid & 127) << 2) + g;

    const float* xb = x + (size_t)b * N_ * D_;
    const float* at = att_t + (size_t)b * N_ * N_;

    float a = 0.f;
    for (int j = 0; j < N_; ++j)
        a += at[(size_t)j * N_ + i] * xb[j * D_ + c];
    agg[g][c] = a;
    xr[g][c]  = xb[i * D_ + c];
    __syncthreads();

    float hv = b1[c] + b2[c];
    for (int d = 0; d < 64; ++d)
        hv += agg[g][d] * W1[d * 64 + c] + xr[g][d] * W2[d * 64 + c];
    h[((size_t)b * N_ + i) * D_ + c] = hv;
}

// ---------------------------------------------------------------------------
// K3b: deterministic batch-norm statistics over all B_*N_ rows (single block).
// ---------------------------------------------------------------------------
__global__ __launch_bounds__(256)
void k3b_stats(const float* __restrict__ h, float* __restrict__ stats)
{
    __shared__ float s1[256], s2l[256];
    const int t = threadIdx.x;
    const int c = t & 63, g = t >> 6;
    float s = 0.f, s2 = 0.f;
    for (int row = g; row < B_ * N_; row += 4) {
        float v = h[(size_t)row * 64 + c];
        s += v; s2 += v * v;
    }
    s1[t] = s; s2l[t] = s2; __syncthreads();
    if (g == 0) {
        for (int gg = 1; gg < 4; ++gg) { s += s1[gg * 64 + c]; s2 += s2l[gg * 64 + c]; }
        const float invn = 1.f / (float)(B_ * N_);
        float mean = s * invn;
        float var  = s2 * invn - mean * mean;
        stats[c]      = mean;
        stats[64 + c] = rsqrtf(var + 1e-5f);
    }
}

// ---------------------------------------------------------------------------
// K4: BN + SELU + sigmoid pool score + exact top-k (rank counting, jax
// top_k tie-breaking) + gather hw = h*pscore. One block per batch.
// ---------------------------------------------------------------------------
__global__ __launch_bounds__(512)
void k4_pool(const float* __restrict__ h, const float* __restrict__ stats,
             const float* __restrict__ gamma, const float* __restrict__ beta,
             const float* __restrict__ poolw, const float* __restrict__ poolb,
             float* __restrict__ out)
{
    __shared__ float mean[64], rstd[64], gm[64], bt[64], pw[64];
    __shared__ float ps[N_];
    __shared__ int   nodeOf[K_];
    const int t = threadIdx.x;
    const int b = blockIdx.x;
    if (t < 64) {
        mean[t] = stats[t]; rstd[t] = stats[64 + t];
        gm[t] = gamma[t]; bt[t] = beta[t]; pw[t] = poolw[t];
    }
    __syncthreads();

    const float* hb = h + (size_t)b * N_ * D_;
    float acc = poolb[0];
    for (int c = 0; c < 64; ++c) {
        float z = (hb[t * 64 + c] - mean[c]) * rstd[c] * gm[c] + bt[c];
        acc += selu_f(z) * pw[c];
    }
    const float s = 1.f / (1.f + __expf(-acc));
    ps[t] = s; __syncthreads();

    int rank = 0;
    for (int j = 0; j < N_; ++j) {
        float sj = ps[j];
        rank += (sj > s) || (sj == s && j < t);
    }
    if (rank < K_) nodeOf[rank] = t;
    __syncthreads();

    for (int e = t; e < K_ * 64; e += 512) {
        int r = e >> 6, c = e & 63;
        int n = nodeOf[r];
        float z = (hb[n * 64 + c] - mean[c]) * rstd[c] * gm[c] + bt[c];
        out[((size_t)b * K_ + r) * 64 + c] = selu_f(z) * ps[n];
    }
}

// ---------------------------------------------------------------------------
extern "C" void kernel_launch(void* const* d_in, const int* in_sizes, int n_in,
                              void* d_out, int out_size, void* d_ws, size_t ws_size,
                              hipStream_t stream)
{
    const float* x     = (const float*)d_in[0];
    const float* Wa    = (const float*)d_in[1];   // att_proj_w (64,64)
    const float* ba    = (const float*)d_in[2];   // att_proj_b (64)
    const float* attw  = (const float*)d_in[3];   // att_weight (64,1)
    const float* W1    = (const float*)d_in[4];   // proj_with_att_w
    const float* b1    = (const float*)d_in[5];
    const float* W2    = (const float*)d_in[6];   // proj_without_att_w
    const float* b2    = (const float*)d_in[7];
    const float* gamma = (const float*)d_in[8];
    const float* beta  = (const float*)d_in[9];
    const float* poolw = (const float*)d_in[10];
    const float* poolb = (const float*)d_in[11];
    float* out = (float*)d_out;

    float* scores = (float*)d_ws;                          // B*N*N  = 4 MB
    float* hbuf   = scores + (size_t)B_ * N_ * N_;         // B*N*64 = 512 KB
    float* stats  = hbuf + (size_t)B_ * N_ * D_;           // 128 floats

    k1_scores  <<<B_ * 32 * 8, 128, 0, stream>>>(x, Wa, ba, attw, scores);
    k2_softmax <<<B_ * N_,     128, 0, stream>>>(scores);
    k3_agg_h   <<<B_ * 128,    256, 0, stream>>>(x, scores, W1, b1, W2, b2, hbuf);
    k3b_stats  <<<1,           256, 0, stream>>>(hbuf, stats);
    k4_pool    <<<B_,          512, 0, stream>>>(hbuf, stats, gamma, beta, poolw, poolb, out);
}
